// model_39908836114746
// MI455X (gfx1250) — matrix operand order repaired and checked by reading, not compiled, not run on hardware
//
#include <hip/hip_runtime.h>
#include <hip/hip_bf16.h>

typedef __attribute__((ext_vector_type(16))) _Float16 v16h;
typedef __attribute__((ext_vector_type(8)))  float    v8f;

#if defined(__HIP_DEVICE_COMPILE__) && __has_builtin(__builtin_amdgcn_global_load_async_to_lds_b128)
#define USE_ASYNC_LDS 1
#else
#define USE_ASYNC_LDS 0
#endif

typedef int vi4 __attribute__((vector_size(4 * sizeof(int))));
typedef __attribute__((address_space(1))) vi4 gvi4;
typedef __attribute__((address_space(3))) vi4 lvi4;

__device__ __forceinline__ void async_copy16(void* lds_dst, const void* gsrc) {
#if USE_ASYNC_LDS
    __builtin_amdgcn_global_load_async_to_lds_b128(
        (gvi4*)const_cast<void*>(gsrc), (lvi4*)lds_dst, 0, 0);
#endif
}

__device__ __forceinline__ void wait_async() {
#if USE_ASYNC_LDS
#if __has_builtin(__builtin_amdgcn_s_wait_asynccnt)
    __builtin_amdgcn_s_wait_asynccnt(0);
#else
    asm volatile("s_wait_asynccnt 0x0" ::: "memory");
#endif
#endif
}

__device__ __forceinline__ v8f zero8() {
    v8f z;
#pragma unroll
    for (int e = 0; e < 8; ++e) z[e] = 0.0f;
    return z;
}

// 16-bit A-matrix 16x32 per-lane K pattern (ISA 7.12.2):
// lanes 0-15: j0..7 -> K0..7,  j8..15 -> K16..23
// lanes16-31: j0..7 -> K8..15, j8..15 -> K24..31
__device__ __forceinline__ int a_kpat(int j, int hi) {
    return (j < 8) ? (j + hi * 8) : (j + 8 + hi * 8);
}
// 16-bit B-matrix 32x16 per-lane K pattern (same as A):
// lanes 0-15: j0..7 -> K0..7, j8..15 -> K16..23; lanes 16-31: K8..15, K24..31
__device__ __forceinline__ int b_kpat(int j, int hi) {
    return (j < 8) ? (j + hi * 8) : (j + 8 + hi * 8);
}

__device__ __forceinline__ v8f wmma_f16(v16h a, v16h b, v8f c) {
    return __builtin_amdgcn_wmma_f32_16x16x32_f16(false, a, false, b, (short)0, c,
                                                  false, false);
}

// ---------------------------------------------------------------------------
// Kernel 1: conv1 as GEMM  M=12544 (bt*49), N=512, K=8192 (c*4 + ky*2 + kx)
// A = im2col(video) fp32 -> f16 on the fly; B rows = conv1_w rows (contiguous)
// Epilogue: +bias, ReLU, store f16 row-major X1[M][512]
// ---------------------------------------------------------------------------
__global__ __launch_bounds__(256) void conv1_gemm(const float* __restrict__ video,
                                                  const float* __restrict__ w,
                                                  const float* __restrict__ bias,
                                                  _Float16* __restrict__ X1) {
    __shared__ __align__(32) _Float16 Als[8 * 32 * 16];  // 8 m-tiles, frag order
    __shared__ __align__(32) _Float16 Bls[8 * 32 * 16];  // 8 n-tiles, frag order

    const int tid = threadIdx.x;
    const int lane = tid & 31, wave = tid >> 5;
    const int l = lane & 15, hi = lane >> 4;
    const int bm = blockIdx.x * 128, bn = blockIdx.y * 128;

    // fill identity: this thread owns fragment (tile fmt, lane) of both LDS tiles
    const int fmt = tid >> 5;
    const int arow = bm + fmt * 16 + l;               // global M row
    const int bt = arow / 49, p = arow % 49;
    const int oy = p / 7, ox = p % 7;
    const long abase = (long)bt * 401408 + (long)(oy * 2) * 14 + (ox * 2);
    const long bcol = (long)(bn + fmt * 16 + l);      // global N col for B fill

    const int mt0 = (wave >> 1) * 2;                  // 2 m-tiles per wave
    const int nt0 = (wave & 1) * 4;                   // 4 n-tiles per wave

    v8f acc[2][4];
#pragma unroll
    for (int i = 0; i < 2; ++i)
#pragma unroll
        for (int j = 0; j < 4; ++j) acc[i][j] = zero8();

    for (int kt = 0; kt < 256; ++kt) {
        const int kb = kt * 32;
        // advisory prefetch of the strided im2col A stream, 4 k-tiles ahead
        if (kt + 4 < 256) {
            int cf = (kb + 128) >> 2;
            __builtin_prefetch(&video[abase + (long)cf * 196], 0, 1);
        }
        // A tile (im2col gather + cvt)
#pragma unroll
        for (int j = 0; j < 16; ++j) {
            int k = kb + a_kpat(j, hi);
            int c = k >> 2, q = k & 3;
            float v = video[abase + (long)c * 196 + (q >> 1) * 14 + (q & 1)];
            Als[(fmt * 32 + lane) * 16 + j] = (_Float16)v;
        }
        // B tile (contiguous rows of conv1_w)
#pragma unroll
        for (int j = 0; j < 16; ++j) {
            int k = kb + b_kpat(j, hi);
            Bls[(fmt * 32 + lane) * 16 + j] = (_Float16)w[bcol * 8192 + k];
        }
        __syncthreads();

        v16h av[2], bv[4];
#pragma unroll
        for (int i = 0; i < 2; ++i)
            av[i] = *(const v16h*)&Als[((mt0 + i) * 32 + lane) * 16];
#pragma unroll
        for (int j = 0; j < 4; ++j)
            bv[j] = *(const v16h*)&Bls[((nt0 + j) * 32 + lane) * 16];
#pragma unroll
        for (int i = 0; i < 2; ++i)
#pragma unroll
            for (int j = 0; j < 4; ++j) acc[i][j] = wmma_f16(av[i], bv[j], acc[i][j]);
        __syncthreads();
    }

    // epilogue: bias + relu -> f16
#pragma unroll
    for (int i = 0; i < 2; ++i) {
#pragma unroll
        for (int jn = 0; jn < 4; ++jn) {
            int gn = bn + (nt0 + jn) * 16 + l;
            float bz = bias[gn];
#pragma unroll
            for (int v = 0; v < 8; ++v) {
                int gm = bm + (mt0 + i) * 16 + v + hi * 8;
                float x = acc[i][jn][v] + bz;
                x = x > 0.0f ? x : 0.0f;
                X1[(long)gm * 512 + gn] = (_Float16)x;
            }
        }
    }
}

// ---------------------------------------------------------------------------
// Kernel 2: conv2 (1x1) as GEMM  M=12544, N=256, K=512
// A = X1 (f16) moved by the async-to-LDS engine (fragment rows are two
// contiguous 16B runs in global). B = conv2_w. Epilogue: +bias, store
// transposed X2T[frame][c][p]
// ---------------------------------------------------------------------------
__global__ __launch_bounds__(256) void conv2_gemm(const _Float16* __restrict__ X1,
                                                  const float* __restrict__ w,
                                                  const float* __restrict__ bias,
                                                  float* __restrict__ X2T) {
    __shared__ __align__(32) _Float16 Als[8 * 32 * 16];
    __shared__ __align__(32) _Float16 Bls[8 * 32 * 16];

    const int tid = threadIdx.x;
    const int lane = tid & 31, wave = tid >> 5;
    const int l = lane & 15, hi = lane >> 4;
    const int bm = blockIdx.x * 128, bn = blockIdx.y * 128;

    const int fmt = tid >> 5;
    const long arow = (long)(bm + fmt * 16 + l);
    const long bcol = (long)(bn + fmt * 16 + l);

    const int mt0 = (wave >> 1) * 2;
    const int nt0 = (wave & 1) * 4;

    v8f acc[2][4];
#pragma unroll
    for (int i = 0; i < 2; ++i)
#pragma unroll
        for (int j = 0; j < 4; ++j) acc[i][j] = zero8();

    for (int kt = 0; kt < 16; ++kt) {
        const int kb = kt * 32;
#if USE_ASYNC_LDS
        {
            // lane's fragment = two contiguous 8-half (16B) runs: k=kb+hi*8.. and kb+16+hi*8..
            const _Float16* g0 = X1 + arow * 512 + kb + hi * 8;
            _Float16* d0 = &Als[(fmt * 32 + lane) * 16];
            async_copy16(d0, g0);
            async_copy16(d0 + 8, g0 + 16);
        }
#else
#pragma unroll
        for (int j = 0; j < 16; ++j) {
            int k = kb + a_kpat(j, hi);
            Als[(fmt * 32 + lane) * 16 + j] = X1[arow * 512 + k];
        }
#endif
#pragma unroll
        for (int j = 0; j < 16; ++j) {
            int k = kb + b_kpat(j, hi);
            Bls[(fmt * 32 + lane) * 16 + j] = (_Float16)w[bcol * 512 + k];
        }
        wait_async();
        __syncthreads();

        v16h av[2], bv[4];
#pragma unroll
        for (int i = 0; i < 2; ++i)
            av[i] = *(const v16h*)&Als[((mt0 + i) * 32 + lane) * 16];
#pragma unroll
        for (int j = 0; j < 4; ++j)
            bv[j] = *(const v16h*)&Bls[((nt0 + j) * 32 + lane) * 16];
#pragma unroll
        for (int i = 0; i < 2; ++i)
#pragma unroll
            for (int j = 0; j < 4; ++j) acc[i][j] = wmma_f16(av[i], bv[j], acc[i][j]);
        __syncthreads();
    }

#pragma unroll
    for (int i = 0; i < 2; ++i) {
#pragma unroll
        for (int jn = 0; jn < 4; ++jn) {
            int gn = bn + (nt0 + jn) * 16 + l;
            float bz = bias[gn];
#pragma unroll
            for (int v = 0; v < 8; ++v) {
                int gm = bm + (mt0 + i) * 16 + v + hi * 8;
                int frame = gm / 49, pp = gm % 49;
                X2T[((long)frame * 256 + gn) * 49 + pp] = acc[i][jn][v] + bz;
            }
        }
    }
}

// ---------------------------------------------------------------------------
// Kernel 3: per frame: avg+max pool -> LayerNorm(xln) -> giax (x-part of att
// GRU input gates, t-parallel) and xa projection [49][32]
// ---------------------------------------------------------------------------
__global__ __launch_bounds__(256) void frame_stats(
    const float* __restrict__ X2T, const float* __restrict__ att_Wih,
    const float* __restrict__ att_bih, const float* __restrict__ xa_W,
    const float* __restrict__ xa_b, const float* __restrict__ xln_g,
    const float* __restrict__ xln_b, float* __restrict__ giax,
    float* __restrict__ xa_out) {
    __shared__ __align__(16) float Xls[256 * 49];
    __shared__ float sbuf[256];
    __shared__ float xlnb[256];
    __shared__ float mv[2];

    const int fr = blockIdx.x;
    const int tid = threadIdx.x;
    const float* src = X2T + (long)fr * 12544;
#if USE_ASYNC_LDS
    // 12544 floats = 3136 x 16B chunks, straight async copy into LDS
    for (int ch = tid; ch < 3136; ch += 256) async_copy16(&Xls[ch * 4], src + ch * 4);
    wait_async();
#else
    for (int i = tid; i < 12544; i += 256) Xls[i] = src[i];
#endif
    __syncthreads();

    {  // avg + max over P for channel tid
        float s = 0.0f, mx = -3.0e38f;
        for (int p = 0; p < 49; ++p) {
            float v = Xls[tid * 49 + p];
            s += v;
            mx = v > mx ? v : mx;
        }
        sbuf[tid] = s * (1.0f / 49.0f) + mx;
    }
    __syncthreads();
    if (tid == 0) {  // LN stats over 256 channels
        float m = 0.0f;
        for (int i = 0; i < 256; ++i) m += sbuf[i];
        m *= (1.0f / 256.0f);
        float v = 0.0f;
        for (int i = 0; i < 256; ++i) {
            float d = sbuf[i] - m;
            v += d * d;
        }
        v *= (1.0f / 256.0f);
        mv[0] = m;
        mv[1] = rsqrtf(v + 1e-5f);
    }
    __syncthreads();
    xlnb[tid] = (sbuf[tid] - mv[0]) * mv[1] * xln_g[tid] + xln_b[tid];
    __syncthreads();

    // giax[fr][n] = att_bih[n] + sum_c xln[c] * att_Wih[n][c]  (c in [0,256))
    for (int n = tid; n < 384; n += 256) {
        float g = att_bih[n];
        const float* wr = att_Wih + (long)n * 384;
        for (int c = 0; c < 256; ++c) g += xlnb[c] * wr[c];
        giax[(long)fr * 384 + n] = g;
    }
    // xa[fr][p][a] = xa_b[a] + sum_c X[c][p] * xa_W[a][c]
    for (int idx = tid; idx < 1568; idx += 256) {
        int p = idx >> 5, a = idx & 31;
        float v = xa_b[a];
        const float* wr = xa_W + a * 256;
        for (int c = 0; c < 256; ++c) v += Xls[c * 49 + p] * wr[c];
        xa_out[((long)fr * 49 + p) * 32 + a] = v;
    }
}

// ---------------------------------------------------------------------------
// Kernel 4: sequential scan over T=32. One workgroup, 8 waves.
// Per-step 8x{128,256} @ 384 matmuls on WMMA (M padded to 16), rest on VALU.
// ---------------------------------------------------------------------------
__global__ __launch_bounds__(256) void scan_kernel(
    const float* __restrict__ h_top0, const float* __restrict__ h_att0,
    const float* __restrict__ att_Wih, const float* __restrict__ att_Whh,
    const float* __restrict__ att_bhh, const float* __restrict__ top_Wih,
    const float* __restrict__ top_Whh, const float* __restrict__ top_bih,
    const float* __restrict__ top_bhh, const float* __restrict__ ha_W,
    const float* __restrict__ ha_b, const float* __restrict__ a_W,
    const float* __restrict__ score_W, const float* __restrict__ score_b,
    const float* __restrict__ hln_g, const float* __restrict__ hln_b,
    const float* __restrict__ giax, const float* __restrict__ xa,
    const float* __restrict__ X2T, float* __restrict__ out) {
    __shared__ float h_top[8 * 128], h_att[8 * 128], lnh[8 * 128];
    __shared__ float gi[8 * 384], gh[8 * 384];
    __shared__ float hav[8 * 32], logits[8 * 49], alpha[8 * 49];
    __shared__ float top_in[8 * 256];
    __shared__ float mv[16];
    __shared__ __align__(32) _Float16 Af[12 * 32 * 16];  // fragment buffers

    const int tid = threadIdx.x;
    const int lane = tid & 31, wave = tid >> 5;
    const int l = lane & 15, hi = lane >> 4;

    for (int i = tid; i < 1024; i += 256) {
        h_top[i] = h_top0[i];
        h_att[i] = h_att0[i];
    }
    __syncthreads();

    for (int t = 0; t < 32; ++t) {
        // ---- LN(h_top) ----
        if (tid < 8) {
            float m = 0.0f;
            for (int k = 0; k < 128; ++k) m += h_top[tid * 128 + k];
            m *= (1.0f / 128.0f);
            float v = 0.0f;
            for (int k = 0; k < 128; ++k) {
                float d = h_top[tid * 128 + k] - m;
                v += d * d;
            }
            mv[tid * 2] = m;
            mv[tid * 2 + 1] = rsqrtf(v * (1.0f / 128.0f) + 1e-5f);
        }
        __syncthreads();
        for (int i = tid; i < 1024; i += 256) {
            int b = i >> 7, k = i & 127;
            lnh[i] = (h_top[i] - mv[b * 2]) * mv[b * 2 + 1] * hln_g[k] + hln_b[k];
        }
        __syncthreads();
        // ---- build A fragments: Af[0..3]=lnh (K=128), Af[4..7]=h_att (K=128)
        {
            int which = tid >> 7;  // 0: lnh, 1: h_att
            int loc = tid & 127;
            int kt2 = loc >> 5, ln2 = loc & 31;
            int l2 = ln2 & 15, hi2 = ln2 >> 4;
            const float* srcm = which ? h_att : lnh;
            _Float16* dst = &Af[((which * 4 + kt2) * 32 + ln2) * 16];
#pragma unroll
            for (int j = 0; j < 16; ++j) {
                int k = kt2 * 32 + a_kpat(j, hi2);
                dst[j] = (l2 < 8) ? (_Float16)srcm[l2 * 128 + k] : (_Float16)0.0f;
            }
        }
        __syncthreads();
        // ---- phase A WMMA: gi += lnh@Wih_h^T (+giax), gh = h_att@Whh^T (+bhh)
        for (int q = 0; q < 6; ++q) {
            int job = wave * 6 + q;
            int mm = job / 24;      // 0: gi, 1: gh
            int ntile = job % 24;   // N tile of 384
            int n = ntile * 16 + l;
            v8f acc = zero8();
            for (int kt2 = 0; kt2 < 4; ++kt2) {
                v16h av = *(const v16h*)&Af[((mm * 4 + kt2) * 32 + lane) * 16];
                v16h bv;
#pragma unroll
                for (int j = 0; j < 16; ++j) {
                    int k = kt2 * 32 + b_kpat(j, hi);
                    float wv = (mm == 0) ? att_Wih[(long)n * 384 + 256 + k]
                                         : att_Whh[(long)n * 128 + k];
                    bv[j] = (_Float16)wv;
                }
                acc = wmma_f16(av, bv, acc);
            }
            if (hi == 0) {  // rows 0..7 (batch) only
#pragma unroll
                for (int v = 0; v < 8; ++v) {
                    int col = ntile * 16 + l;
                    if (mm == 0)
                        gi[v * 384 + col] = acc[v] + giax[((long)v * 32 + t) * 384 + col];
                    else
                        gh[v * 384 + col] = acc[v] + att_bhh[col];
                }
            }
        }
        __syncthreads();
        // ---- attention GRU elementwise ----
        for (int i = tid; i < 1024; i += 256) {
            int b = i >> 7, k = i & 127;
            float ir = gi[b * 384 + k], iz = gi[b * 384 + 128 + k], in = gi[b * 384 + 256 + k];
            float hr = gh[b * 384 + k], hz = gh[b * 384 + 128 + k], hn = gh[b * 384 + 256 + k];
            float r = 1.0f / (1.0f + __expf(-(ir + hr)));
            float z = 1.0f / (1.0f + __expf(-(iz + hz)));
            float nn = tanhf(in + r * hn);
            h_att[i] = (1.0f - z) * nn + z * h_att[i];
        }
        __syncthreads();
        // ---- ha = h_att @ ha_W^T + ha_b ----
        {
            int b = tid >> 5, a = tid & 31;
            float v = ha_b[a];
            const float* wr = ha_W + a * 128;
            for (int k = 0; k < 128; ++k) v += h_att[b * 128 + k] * wr[k];
            hav[tid] = v;
        }
        __syncthreads();
        // ---- logits ----
        for (int i = tid; i < 392; i += 256) {
            int b = i / 49, p = i % 49;
            const float* xr = xa + (((long)(b * 32 + t)) * 49 + p) * 32;
            float s = 0.0f;
            for (int a = 0; a < 32; ++a) s += a_W[a] * tanhf(xr[a] + hav[b * 32 + a]);
            logits[i] = s;
        }
        __syncthreads();
        // ---- softmax over P, emit soft_att ----
        if (tid < 8) {
            int b = tid;
            float mx = -3.0e38f;
            for (int p = 0; p < 49; ++p) mx = logits[b * 49 + p] > mx ? logits[b * 49 + p] : mx;
            float s = 0.0f;
            for (int p = 0; p < 49; ++p) {
                float e = __expf(logits[b * 49 + p] - mx);
                alpha[b * 49 + p] = e;
                s += e;
            }
            float inv = 1.0f / s;
            for (int p = 0; p < 49; ++p) {
                float al = alpha[b * 49 + p] * inv;
                alpha[b * 49 + p] = al;
                out[8 + ((long)b * 32 + t) * 49 + p] = al;
            }
        }
        __syncthreads();
        // ---- top_in = Xf_t @ alpha ----
        for (int i = tid; i < 2048; i += 256) {
            int b = i >> 8, c = i & 255;
            const float* xr = X2T + (((long)(b * 32 + t)) * 256 + c) * 49;
            float s = 0.0f;
            for (int p = 0; p < 49; ++p) s += xr[p] * alpha[b * 49 + p];
            top_in[i] = s;
        }
        __syncthreads();
        // ---- build Af[0..7]=top_in (K=256), Af[8..11]=h_top (K=128) ----
        for (int idx = tid; idx < 384; idx += 256) {
            int kt2, ln2, K, buf;
            const float* srcm;
            if (idx < 256) {
                kt2 = idx >> 5; ln2 = idx & 31; srcm = top_in; K = 256; buf = kt2;
            } else {
                int ix = idx - 256;
                kt2 = ix >> 5; ln2 = ix & 31; srcm = h_top; K = 128; buf = 8 + kt2;
            }
            int l2 = ln2 & 15, hi2 = ln2 >> 4;
            _Float16* dst = &Af[(buf * 32 + ln2) * 16];
#pragma unroll
            for (int j = 0; j < 16; ++j) {
                int k = kt2 * 32 + a_kpat(j, hi2);
                dst[j] = (l2 < 8) ? (_Float16)srcm[l2 * K + k] : (_Float16)0.0f;
            }
        }
        __syncthreads();
        // ---- phase B WMMA: gi = top_in@Wih^T + bih ; gh = h_top@Whh^T + bhh
        for (int q = 0; q < 6; ++q) {
            int job = wave * 6 + q;
            int mm = job / 24;
            int ntile = job % 24;
            int n = ntile * 16 + l;
            v8f acc = zero8();
            int nkt = (mm == 0) ? 8 : 4;
            int base = (mm == 0) ? 0 : 8;
            for (int kt2 = 0; kt2 < nkt; ++kt2) {
                v16h av = *(const v16h*)&Af[((base + kt2) * 32 + lane) * 16];
                v16h bv;
#pragma unroll
                for (int j = 0; j < 16; ++j) {
                    int k = kt2 * 32 + b_kpat(j, hi);
                    float wv = (mm == 0) ? top_Wih[(long)n * 256 + k]
                                         : top_Whh[(long)n * 128 + k];
                    bv[j] = (_Float16)wv;
                }
                acc = wmma_f16(av, bv, acc);
            }
            if (hi == 0) {
#pragma unroll
                for (int v = 0; v < 8; ++v) {
                    int col = ntile * 16 + l;
                    if (mm == 0)
                        gi[v * 384 + col] = acc[v] + top_bih[col];
                    else
                        gh[v * 384 + col] = acc[v] + top_bhh[col];
                }
            }
        }
        __syncthreads();
        // ---- top GRU elementwise ----
        for (int i = tid; i < 1024; i += 256) {
            int b = i >> 7, k = i & 127;
            float ir = gi[b * 384 + k], iz = gi[b * 384 + 128 + k], in = gi[b * 384 + 256 + k];
            float hr = gh[b * 384 + k], hz = gh[b * 384 + 128 + k], hn = gh[b * 384 + 256 + k];
            float r = 1.0f / (1.0f + __expf(-(ir + hr)));
            float z = 1.0f / (1.0f + __expf(-(iz + hz)));
            float nn = tanhf(in + r * hn);
            h_top[i] = (1.0f - z) * nn + z * h_top[i];
        }
        __syncthreads();
    }
    // ---- final score ----
    if (tid < 8) {
        float s = score_b[0];
        for (int k = 0; k < 128; ++k) s += h_top[tid * 128 + k] * score_W[k];
        out[tid] = s;
    }
}

// ---------------------------------------------------------------------------
extern "C" void kernel_launch(void* const* d_in, const int* in_sizes, int n_in,
                              void* d_out, int out_size, void* d_ws, size_t ws_size,
                              hipStream_t stream) {
    const float* video   = (const float*)d_in[0];
    const float* h_top0  = (const float*)d_in[1];
    const float* h_att0  = (const float*)d_in[2];
    const float* conv1_w = (const float*)d_in[3];
    const float* conv1_b = (const float*)d_in[4];
    const float* conv2_w = (const float*)d_in[5];
    const float* conv2_b = (const float*)d_in[6];
    const float* att_Wih = (const float*)d_in[7];
    const float* att_Whh = (const float*)d_in[8];
    const float* att_bih = (const float*)d_in[9];
    const float* att_bhh = (const float*)d_in[10];
    const float* top_Wih = (const float*)d_in[11];
    const float* top_Whh = (const float*)d_in[12];
    const float* top_bih = (const float*)d_in[13];
    const float* top_bhh = (const float*)d_in[14];
    const float* xa_W    = (const float*)d_in[15];
    const float* xa_b    = (const float*)d_in[16];
    const float* ha_W    = (const float*)d_in[17];
    const float* ha_b    = (const float*)d_in[18];
    const float* a_W     = (const float*)d_in[19];
    const float* score_W = (const float*)d_in[20];
    const float* score_b = (const float*)d_in[21];
    const float* xln_g   = (const float*)d_in[22];
    const float* xln_b   = (const float*)d_in[23];
    const float* hln_g   = (const float*)d_in[24];
    const float* hln_b   = (const float*)d_in[25];

    char* ws = (char*)d_ws;
    _Float16* X1  = (_Float16*)(ws + 0);            // 12544*512*2  = 12,845,056 B
    float*    X2T = (float*)(ws + 12845056);        // 256*256*49*4 = 12,845,056 B
    float*    giax = (float*)(ws + 25690112);       // 256*384*4    =    393,216 B
    float*    xav  = (float*)(ws + 26083328);       // 256*49*32*4  =  1,605,632 B

    conv1_gemm<<<dim3(98, 4), 256, 0, stream>>>(video, conv1_w, conv1_b, X1);
    conv2_gemm<<<dim3(98, 2), 256, 0, stream>>>(X1, conv2_w, conv2_b, X2T);
    frame_stats<<<256, 256, 0, stream>>>(X2T, att_Wih, att_bih, xa_W, xa_b,
                                         xln_g, xln_b, giax, xav);
    scan_kernel<<<1, 256, 0, stream>>>(h_top0, h_att0, att_Wih, att_Whh, att_bhh,
                                       top_Wih, top_Whh, top_bih, top_bhh,
                                       ha_W, ha_b, a_W, score_W, score_b,
                                       hln_g, hln_b, giax, xav, X2T, (float*)d_out);
}
